// MixtureOfExperts_64639257805113
// MI455X (gfx1250) — compile-verified, matrix operand order checked by reading, not run on hardware
//
#include <hip/hip_runtime.h>
#include <hip/hip_bf16.h>
#include <math.h>
#include <stdint.h>

// ---------------- problem constants ----------------
#define T_TOK 4096          // B*S
#define DDIM  1024
#define HDIM  4096
#define ODIM  1024
#define NEXP  8
#define TOPK  2
#define CAP   (T_TOK * TOPK)   // worst-case assignments per expert
#define MT    32               // assignments (rows) per FFN block

// workspace layout (bytes) -- total ~169 MiB
#define OFF_CNT   0                                     // 8 ints (expert counters)
#define OFF_TK    1024                                  // 8*CAP ints       (256 KB)
#define OFF_GATE  (OFF_TK   + NEXP*CAP*4)               // 8*CAP floats     (256 KB)
#define OFF_BW1   (OFF_GATE + NEXP*CAP*4)               // bf16 W1, swizzled (64 MB)
#define OFF_BW2   (OFF_BW1  + (size_t)NEXP*DDIM*HDIM*2) // bf16 W2, swizzled (64 MB)
#define OFF_YBUF  (OFF_BW2  + (size_t)NEXP*HDIM*ODIM*2) // T*2*O floats     (32 MB)
#define OFF_BX    (OFF_YBUF + (size_t)T_TOK*TOPK*ODIM*4) // bf16 x            (8 MB)

typedef __attribute__((ext_vector_type(16))) __bf16 v16bf;
typedef __attribute__((ext_vector_type(4)))  __bf16 v4bf;
typedef __attribute__((ext_vector_type(2)))  __bf16 bf2;
typedef __attribute__((ext_vector_type(8)))  float  v8f;
typedef __attribute__((ext_vector_type(4)))  float  v4f;

// K index of element pair v (0..7) inside a 16-bit WMMA operand vector,
// per CDNA5 ISA 7.12.2 (16-bit operand 16x32): lanes 0-15 half=0, 16-31 half=1.
__device__ __forceinline__ int kpair(int v, int half) {
    return (v < 4) ? (half * 8 + 2 * v) : (16 + half * 8 + 2 * (v - 4));
}

// Load a 16x32 bf16 A-fragment (M=lane%16, K-half=lane/16) from LDS row-major.
__device__ __forceinline__ v16bf load_a_lds(const __bf16* s, int ld, int m, int half, int kofs) {
    v16bf a;
#pragma unroll
    for (int v = 0; v < 8; ++v) {
        bf2 p = *(const bf2*)(s + m * ld + kofs + kpair(v, half));
        a[2 * v]     = p[0];
        a[2 * v + 1] = p[1];
    }
    return a;
}

// Async gather: copy 16 bytes global -> LDS, tracked by ASYNCcnt.
// ISA 15.18.3 opcode 98 (GLOBAL_LOAD_ASYNC_TO_LDS_B128), GV addressing:
//   vdst VGPR holds the LDS byte address.
__device__ __forceinline__ void async_copy16(const void* gsrc, void* ldst) {
    unsigned lds_off = (unsigned)(uintptr_t)ldst;            // low 32 bits = LDS addr
    unsigned long long ga = (unsigned long long)(uintptr_t)gsrc;
    asm volatile("global_load_async_to_lds_b128 %0, %1, off"
                 :: "v"(lds_off), "v"(ga) : "memory");
}

// ---------------- gating: logits -> top2 -> softmax -> scatter ----------------
__global__ void moe_gate_64639257805113(const float* __restrict__ x,
                                        const float* __restrict__ Wg,
                                        const float* __restrict__ bg,
                                        int* __restrict__ cnt,
                                        int* __restrict__ tk_list,
                                        float* __restrict__ gate_list) {
    int lane = threadIdx.x & 31;
    int wv   = threadIdx.x >> 5;
    int t    = blockIdx.x * 8 + wv;
    if (t >= T_TOK) return;

    float acc[NEXP];
#pragma unroll
    for (int e = 0; e < NEXP; ++e) acc[e] = 0.f;

    const float* xr = x + (size_t)t * DDIM;
    for (int d = lane; d < DDIM; d += 32) {
        float xv = xr[d];
#pragma unroll
        for (int e = 0; e < NEXP; ++e) acc[e] += xv * Wg[e * DDIM + d];
    }
#pragma unroll
    for (int e = 0; e < NEXP; ++e) {
        for (int off = 16; off > 0; off >>= 1)
            acc[e] += __shfl_xor(acc[e], off, 32);
    }
    if (lane == 0) {
        float l[NEXP];
#pragma unroll
        for (int e = 0; e < NEXP; ++e) l[e] = acc[e] + bg[e];
        int i0 = 0; float v0 = l[0];
#pragma unroll
        for (int e = 1; e < NEXP; ++e) if (l[e] > v0) { v0 = l[e]; i0 = e; }
        int i1 = -1; float v1 = -3.4e38f;
#pragma unroll
        for (int e = 0; e < NEXP; ++e) if (e != i0 && l[e] > v1) { v1 = l[e]; i1 = e; }
        // softmax over the two selected logits (v0 >= v1)
        float e1 = expf(v1 - v0);
        float s  = 1.f + e1;
        float g0 = 1.f / s, g1 = e1 / s;
        int s0 = atomicAdd(&cnt[i0], 1);
        tk_list[i0 * CAP + s0]   = t * 2 + 0;
        gate_list[i0 * CAP + s0] = g0;
        int s1 = atomicAdd(&cnt[i1], 1);
        tk_list[i1 * CAP + s1]   = t * 2 + 1;
        gate_list[i1 * CAP + s1] = g1;
    }
}

// ---------------- x pre-conversion: fp32 [T,D] -> bf16 [T,D] ------------------
__global__ void moe_cvtx_64639257805113(const float* __restrict__ x,
                                        __bf16* __restrict__ xbf) {
    int i = blockIdx.x * blockDim.x + threadIdx.x;          // float4 index
    if (i >= T_TOK * DDIM / 4) return;
    v4f f = ((const v4f*)x)[i];
    v4bf o;
    o[0] = (__bf16)f[0]; o[1] = (__bf16)f[1];
    o[2] = (__bf16)f[2]; o[3] = (__bf16)f[3];
    ((v4bf*)xbf)[i] = o;
}

// ---------------- weight pre-conversion: fp32 [E,K,N] -> bf16 B-fragments ------
// Fragment (kt,nt) covers k in [kt*32,+32), n in [nt*16,+16). Lane L of a frag:
// n = nt*16 + L%16, K-half = L/16; its 16 elements are stored contiguously so
// the FFN loads each B operand as one 32-byte vector (2x global_load_b128).
__global__ void __launch_bounds__(256)
moe_cvtw_64639257805113(const float* __restrict__ src, __bf16* __restrict__ dst,
                        int K, int N, int nKT, int nNT) {
    int lane = threadIdx.x & 31;
    int wv   = threadIdx.x >> 5;
    long fragIdx = (long)blockIdx.x * 8 + wv;
    int fpe = nKT * nNT;                       // fragments per expert
    int e   = (int)(fragIdx / fpe);
    int f   = (int)(fragIdx % fpe);
    int kt  = f / nNT, nt = f % nNT;

    const float* srcE = src + (size_t)e * K * N;
    v16bf* dstE = (v16bf*)dst + (size_t)e * ((size_t)K * N / 16);

    int half = lane >> 4;
    int n    = nt * 16 + (lane & 15);
    v16bf b;
#pragma unroll
    for (int j = 0; j < 16; ++j) {
        int k = kt * 32 + kpair(j >> 1, half) + (j & 1);
        b[j] = (__bf16)srcE[(size_t)k * N + n];
    }
    dstE[((size_t)kt * nNT + nt) * 32 + lane] = b;
}

// ---------------- fused expert FFN: y = relu(x@W1+b1)@W2+b2, scaled by gate ----
__global__ void __launch_bounds__(256)
moe_ffn_64639257805113(const __bf16* __restrict__ xbf,
                       const __bf16* __restrict__ bfW1,
                       const float*  __restrict__ b1,
                       const __bf16* __restrict__ bfW2,
                       const float*  __restrict__ b2,
                       const int*    __restrict__ cnt,
                       const int*    __restrict__ tk_list,
                       const float*  __restrict__ gate_list,
                       float* __restrict__ ybuf) {
    const int e    = blockIdx.y;
    const int tile = blockIdx.x;
    const int nval = cnt[e];
    if (tile * MT >= nval) return;   // whole block exits (no barriers crossed)

    __shared__ __align__(16) __bf16 xs[MT * DDIM];   // 64 KB x-tile (bf16)
    __shared__ __align__(16) __bf16 hs[MT * 256];    // 16 KB h-chunk (bf16)
    __shared__ int   tk_s[MT];
    __shared__ float gate_s[MT];

    const int tid  = threadIdx.x;
    const int lane = tid & 31;
    const int wv   = tid >> 5;      // 0..7
    const int half = lane >> 4;     // K-half for A operands
    const int nn   = lane & 15;     // M row (A) / N col (B,C,D)

    if (tid < MT) {
        int slot = tile * MT + tid;
        bool ok  = slot < nval;
        tk_s[tid]   = ok ? tk_list[e * CAP + slot]   : -1;
        gate_s[tid] = ok ? gate_list[e * CAP + slot] : 0.f;
    }
    __syncthreads();

    // stage x rows into LDS via async copy (gather, 16B per lane, ASYNCcnt).
    // Padded rows alias row 0; their outputs are discarded in the epilogue.
    for (int i = tid; i < MT * (DDIM / 8); i += 256) {
        int row = i >> 7, c = i & (DDIM / 8 - 1);          // 128 x 16B chunks/row
        int tk   = tk_s[row];
        int trow = (tk >= 0) ? (tk >> 1) : 0;
        async_copy16(xbf + (size_t)trow * DDIM + c * 8,
                     xs + row * DDIM + c * 8);
    }
    asm volatile("s_wait_asynccnt 0" ::: "memory");
    __syncthreads();

    const v16bf* W1f = (const v16bf*)bfW1 + (size_t)e * ((size_t)DDIM * HDIM / 16);
    const v16bf* W2f = (const v16bf*)bfW2 + (size_t)e * ((size_t)HDIM * ODIM / 16);
    const float* b1e = b1 + (size_t)e * HDIM;
    const float* b2e = b2 + (size_t)e * ODIM;

    v8f acc[16];   // 32 x 128 f32 output accumulators per wave (cols wv*128..+128)
#pragma unroll
    for (int i = 0; i < 16; ++i) acc[i] = (v8f)0.f;

#pragma unroll 1
    for (int it = 0; it < HDIM / 256; ++it) {
        // ---- layer 1: this wave computes h columns [it*256 + wv*32, +32) x 32 rows
        v8f c0 = (v8f)0.f, c1 = (v8f)0.f, c2 = (v8f)0.f, c3 = (v8f)0.f;
        const int nt0 = it * 16 + wv * 2;   // W1 fragment column tile (16-wide)
#pragma unroll 1
        for (int kk = 0; kk < DDIM / 32; ++kk) {
            v16bf a0 = load_a_lds(xs,             DDIM, nn, half, kk * 32);
            v16bf a1 = load_a_lds(xs + 16 * DDIM, DDIM, nn, half, kk * 32);
            v16bf b0 = W1f[((size_t)kk * (HDIM / 16) + nt0)     * 32 + lane];
            v16bf b1v= W1f[((size_t)kk * (HDIM / 16) + nt0 + 1) * 32 + lane];
            c0 = __builtin_amdgcn_wmma_f32_16x16x32_bf16(false, a0, false, b0,  (short)0, c0, false, false);
            c1 = __builtin_amdgcn_wmma_f32_16x16x32_bf16(false, a0, false, b1v, (short)0, c1, false, false);
            c2 = __builtin_amdgcn_wmma_f32_16x16x32_bf16(false, a1, false, b0,  (short)0, c2, false, false);
            c3 = __builtin_amdgcn_wmma_f32_16x16x32_bf16(false, a1, false, b1v, (short)0, c3, false, false);
        }
        const int h0 = it * 256 + wv * 32;
        float biasA = b1e[h0 + nn];
        float biasB = b1e[h0 + 16 + nn];
#pragma unroll
        for (int r = 0; r < 8; ++r) {
            int m0 = r + 8 * half;                 // C/D layout: m = vgpr + 8*(lane/16)
            int m1 = m0 + 16;
            hs[m0 * 256 + wv * 32 + nn]      = (__bf16)fmaxf(c0[r] + biasA, 0.f);
            hs[m0 * 256 + wv * 32 + 16 + nn] = (__bf16)fmaxf(c1[r] + biasB, 0.f);
            hs[m1 * 256 + wv * 32 + nn]      = (__bf16)fmaxf(c2[r] + biasA, 0.f);
            hs[m1 * 256 + wv * 32 + 16 + nn] = (__bf16)fmaxf(c3[r] + biasB, 0.f);
        }
        __syncthreads();   // full 32x256 h-chunk visible

        // ---- layer 2: consume h-chunk into this wave's 32x128 y slice ----
#pragma unroll 1
        for (int ks = 0; ks < 8; ++ks) {
            v16bf a0 = load_a_lds(hs,            256, nn, half, ks * 32);
            v16bf a1 = load_a_lds(hs + 16 * 256, 256, nn, half, ks * 32);
            int kt = it * 8 + ks;
#pragma unroll
            for (int nt = 0; nt < 8; ++nt) {
                v16bf b = W2f[((size_t)kt * (ODIM / 16) + wv * 8 + nt) * 32 + lane];
                acc[nt]     = __builtin_amdgcn_wmma_f32_16x16x32_bf16(false, a0, false, b, (short)0, acc[nt],     false, false);
                acc[8 + nt] = __builtin_amdgcn_wmma_f32_16x16x32_bf16(false, a1, false, b, (short)0, acc[8 + nt], false, false);
            }
        }
        __syncthreads();   // protect hs before next iteration overwrites
    }

    // ---- epilogue: ybuf[tk, col] = gate * (y + b2) ----
#pragma unroll
    for (int nt = 0; nt < 8; ++nt) {
        int col = wv * 128 + nt * 16 + nn;
        float bb = b2e[col];
#pragma unroll
        for (int mt = 0; mt < 2; ++mt) {
#pragma unroll
            for (int r = 0; r < 8; ++r) {
                int m  = mt * 16 + r + 8 * half;
                int tk = tk_s[m];
                if (tk >= 0)
                    ybuf[(size_t)tk * ODIM + col] = gate_s[m] * (acc[mt * 8 + nt][r] + bb);
            }
        }
    }
}

// ---------------- combine the two expert contributions per token ----------------
__global__ void moe_combine_64639257805113(const float* __restrict__ ybuf,
                                           float* __restrict__ out) {
    int i = blockIdx.x * blockDim.x + threadIdx.x;          // float4 index
    if (i >= T_TOK * ODIM / 4) return;
    size_t t    = (size_t)(i >> 8);                         // 256 float4 per row
    size_t base = (size_t)i + t * (ODIM / 4);               // (t*2)*O/4 + o4
    v4f a = ((const v4f*)ybuf)[base];
    v4f b = ((const v4f*)ybuf)[base + ODIM / 4];
    ((v4f*)out)[i] = a + b;
}

// ---------------- load-balance auxiliary loss ----------------
__global__ void moe_lb_64639257805113(const int* __restrict__ cnt,
                                      float* __restrict__ out) {
    if (threadIdx.x == 0 && blockIdx.x == 0) {
        float s = 0.f;
        float denom = (float)(T_TOK * TOPK) + 1e-8f;
#pragma unroll
        for (int e = 0; e < NEXP; ++e) {
            float f = (float)cnt[e] / denom - 1.0f / NEXP;
            s += f * f;
        }
        out[(size_t)T_TOK * ODIM] = 0.01f * s;
    }
}

extern "C" void kernel_launch(void* const* d_in, const int* in_sizes, int n_in,
                              void* d_out, int out_size, void* d_ws, size_t ws_size,
                              hipStream_t stream) {
    const float* x  = (const float*)d_in[0];
    const float* Wg = (const float*)d_in[1];
    const float* bg = (const float*)d_in[2];
    const float* W1 = (const float*)d_in[3];
    const float* b1 = (const float*)d_in[4];
    const float* W2 = (const float*)d_in[5];
    const float* b2 = (const float*)d_in[6];
    float* out = (float*)d_out;

    char* ws = (char*)d_ws;
    int*    cnt  = (int*)   (ws + OFF_CNT);
    int*    tkl  = (int*)   (ws + OFF_TK);
    float*  gl   = (float*) (ws + OFF_GATE);
    __bf16* bW1  = (__bf16*)(ws + OFF_BW1);
    __bf16* bW2  = (__bf16*)(ws + OFF_BW2);
    float*  ybuf = (float*) (ws + OFF_YBUF);
    __bf16* xbf  = (__bf16*)(ws + OFF_BX);

    (void)hipMemsetAsync(cnt, 0, NEXP * sizeof(int), stream);

    moe_gate_64639257805113<<<T_TOK / 8, 256, 0, stream>>>(x, Wg, bg, cnt, tkl, gl);

    // one-shot bf16 pre-conversions (x plain row-major; weights fragment-swizzled)
    moe_cvtx_64639257805113<<<(T_TOK * DDIM / 4 + 255) / 256, 256, 0, stream>>>(x, xbf);
    {
        int nKT1 = DDIM / 32, nNT1 = HDIM / 16;   // W1: K=D, N=H
        int nKT2 = HDIM / 32, nNT2 = ODIM / 16;   // W2: K=H, N=O
        int blk1 = NEXP * nKT1 * nNT1 / 8;
        int blk2 = NEXP * nKT2 * nNT2 / 8;
        moe_cvtw_64639257805113<<<blk1, 256, 0, stream>>>(W1, bW1, DDIM, HDIM, nKT1, nNT1);
        moe_cvtw_64639257805113<<<blk2, 256, 0, stream>>>(W2, bW2, HDIM, ODIM, nKT2, nNT2);
    }

    dim3 grid(CAP / MT, NEXP);   // worst-case tiles per expert; empty tiles exit
    moe_ffn_64639257805113<<<grid, 256, 0, stream>>>(xbf, bW1, b1, bW2, b2, cnt, tkl, gl, ybuf);

    moe_combine_64639257805113<<<(T_TOK * ODIM / 4 + 255) / 256, 256, 0, stream>>>(ybuf, out);

    moe_lb_64639257805113<<<1, 1, 0, stream>>>(cnt, out);
}
